// SentenceEmbeddingLayer_12987981103233
// MI455X (gfx1250) — matrix operand order repaired and checked by reading, not compiled, not run on hardware
//
#include <hip/hip_runtime.h>
#include <hip/hip_bf16.h>
#include <cstdint>

typedef __attribute__((ext_vector_type(16))) _Float16 v16h;
typedef __attribute__((ext_vector_type(8)))  float    v8f;
typedef __attribute__((ext_vector_type(4)))  unsigned int uint32x4_t_;
typedef __attribute__((ext_vector_type(8)))  int          int32x8_t_;
typedef __attribute__((ext_vector_type(4)))  int          int32x4_t_;

#if __has_builtin(__builtin_amdgcn_tensor_load_to_lds) && \
    __has_builtin(__builtin_amdgcn_s_wait_tensorcnt)
#define USE_TDM 1
#endif

namespace {
constexpr int H_  = 150;   // hidden
constexpr int E_  = 300;   // embed dim
constexpr int S_  = 4096;  // sentence length (char-GRU batch)
constexpr int W_  = 20;    // word length (char-GRU time)
constexpr int N3_ = 450;   // 3*H
constexpr int NP_ = 464;   // N padded to 29 tiles of 16
constexpr int NT_ = 29;    // number of 16-col N tiles
constexpr int KE_ = 320;   // char input K padded (300->320)
constexpr int KH_ = 160;   // hidden K padded (150->160)
constexpr int KS_ = 608;   // sent input K padded (600->608)
}

union HFrag { v16h v; unsigned u[8]; };

__device__ __forceinline__ v8f wmma_f16(v16h a, v16h b, v8f c) {
  return __builtin_amdgcn_wmma_f32_16x16x32_f16(false, a, false, b, (short)0, c,
                                                false, false);
}

__device__ __forceinline__ unsigned ld2h(const _Float16* p) {
  return *reinterpret_cast<const unsigned*>(p);
}

// A-matrix fragment (16x32 f16) from global, row-major, lane holds row M=lane&15.
// ISA layout: VGPR v holds K = (v>>2)*16 + half*8 + (v&3)*2 (+1)
__device__ __forceinline__ v16h load_fragA_g(const _Float16* __restrict__ base,
                                             size_t rowStride, int row16,
                                             int kbase, int lane) {
  HFrag f;
  const _Float16* p = base + (size_t)(row16 + (lane & 15)) * rowStride;
  const int h8 = ((lane >> 4) & 1) << 3;
#pragma unroll
  for (int r = 0; r < 8; ++r) {
    int k = kbase + ((r >> 2) << 4) + h8 + ((r & 3) << 1);
    f.u[r] = ld2h(p + k);
  }
  return f.v;
}

// B-matrix fragment (32x16 f16). B[k][n] = W[n][k] (W row-major over output
// cols). Lane holds col N=lane&15; VGPR v
// holds K = (v>>2)*16 + half*8 + (v&3)*2 (+1).
__device__ __forceinline__ v16h load_fragB_g(const _Float16* __restrict__ base,
                                             size_t rowStride, int col16,
                                             int kbase, int lane) {
  HFrag f;
  const _Float16* p = base + (size_t)(col16 + (lane & 15)) * rowStride;
  const int h8 = ((lane >> 4) & 1) << 3;
#pragma unroll
  for (int r = 0; r < 8; ++r) {
    int k = kbase + ((r >> 2) << 4) + h8 + ((r & 3) << 1);
    f.u[r] = ld2h(p + k);
  }
  return f.v;
}

__device__ __forceinline__ float sigm_(float x) { return 1.f / (1.f + __expf(-x)); }
__device__ __forceinline__ float tanh_(float x) { return 2.f * sigm_(2.f * x) - 1.f; }

// ---------- weight prep: f32 -> padded row-major f16 ----------
__global__ __launch_bounds__(256) void k_pad_w(const float* __restrict__ src,
                                               _Float16* __restrict__ dst,
                                               int N, int K, int Kp, int total) {
  for (int i = blockIdx.x * blockDim.x + threadIdx.x; i < total;
       i += gridDim.x * blockDim.x) {
    int n = i / Kp, k = i % Kp;
    dst[i] = (_Float16)((n < N && k < K) ? src[n * K + k] : 0.f);
  }
}

// sWhh (450x150) -> f16 transposed padded (KH_ x NP_): dst[k][n] = src[n][k]
__global__ __launch_bounds__(256) void k_transpose_w(const float* __restrict__ src,
                                                     _Float16* __restrict__ dst) {
  const int total = KH_ * NP_;
  for (int i = blockIdx.x * blockDim.x + threadIdx.x; i < total;
       i += gridDim.x * blockDim.x) {
    int k = i / NP_, n = i % NP_;
    dst[i] = (_Float16)((k < H_ && n < N3_) ? src[n * H_ + k] : 0.f);
  }
}

// ---------- fused char BiGRU: 16 batch rows per block, 20 steps ----------
__global__ __launch_bounds__(256)
void k_char_bigru(const float* __restrict__ xchars,
                  const _Float16* __restrict__ WihF, const _Float16* __restrict__ WihB,
                  const _Float16* __restrict__ WhhF, const _Float16* __restrict__ WhhB,
                  const float* __restrict__ bihF, const float* __restrict__ bihB,
                  const float* __restrict__ bhhF, const float* __restrict__ bhhB,
                  float* __restrict__ h_char) {
  const int dir = blockIdx.y;
  const _Float16* Wih = dir ? WihB : WihF;
  const _Float16* Whh = dir ? WhhB : WhhF;
  const float* bih = dir ? bihB : bihF;
  const float* bhh = dir ? bhhB : bhhF;
  const int s0 = blockIdx.x * 16;
  const int tid = threadIdx.x;
  const int lane = tid & 31, wave = tid >> 5;

  __shared__ _Float16 Xt[16][KE_];   // input tile (f16)
  __shared__ _Float16 Hh[16][KH_];   // hidden tile (f16, for WMMA)
  __shared__ float    Hf[16][H_];    // hidden tile (f32, exact carry)
  __shared__ float    Gs[16][NP_];   // gx+gh (both biases); also f32 TDM stage
  __shared__ float    Ghn[16][152];  // gh n-third only (with bhh_n)

  float* stage = &Gs[0][0];          // 16*300 f32 = 19200B <= sizeof(Gs)

  for (int i = tid; i < 16 * KH_; i += 256) Hh[i / KH_][i % KH_] = (_Float16)0.f;
  for (int i = tid; i < 16 * H_; i += 256) Hf[i / H_][i % H_] = 0.f;
  __builtin_prefetch(Wih, 0, 3);
  __builtin_prefetch(Whh, 0, 3);
  __syncthreads();

  for (int t = 0; t < W_; ++t) {
    const int tc = dir ? (W_ - 1 - t) : t;
#ifdef USE_TDM
    // Tensor Data Mover: gather 16 rows x 300 f32 (row stride 20*300 elems)
    // into contiguous LDS, tracked by TENSORcnt. Issued once per block.
    if (wave == 0) {
      unsigned ldsOff = (unsigned)(uintptr_t)stage;
      unsigned long long ga =
          (unsigned long long)(uintptr_t)(xchars + ((size_t)s0 * W_ + tc) * E_);
      uint32x4_t_ g0 = {};
      g0[0] = 1u;                                        // count=1, user mode
      g0[1] = ldsOff;                                    // lds_addr
      g0[2] = (unsigned)(ga & 0xffffffffu);              // global_addr[31:0]
      g0[3] = (unsigned)((ga >> 32) & 0x01ffffffu) |     // global_addr[56:32]
              0x80000000u;                               // type=2 ("image")
      int32x8_t_ g1 = {};
      g1[0] = (int)(2u << 16);                           // data_size=4B
      g1[1] = (int)(300u << 16);                         // tensor_dim0 lo16
      g1[2] = (int)(16u << 16);                          // dim0 hi=0 | tensor_dim1=16
      g1[3] = (int)(300u << 16);                         // dim1 hi=0 | tile_dim0=300
      g1[4] = (int)(16u);                                // tile_dim1=16 | tile_dim2=0
      g1[5] = (int)(6000u);                              // tensor_dim0_stride lo32
      g1[6] = 0;                                         // stride hi | dim1_stride lo
      g1[7] = 0;
      int32x4_t_ gz4 = {};
      int32x8_t_ gz8 = {};
      __builtin_amdgcn_tensor_load_to_lds(g0, g1, gz4, gz4, gz8, 0);
      __builtin_amdgcn_s_wait_tensorcnt(0);
    }
    __syncthreads();
    for (int i = tid; i < 16 * KE_; i += 256) {
      int m = i / KE_, c = i % KE_;
      Xt[m][c] = (_Float16)((c < E_) ? stage[m * E_ + c] : 0.f);
    }
    __syncthreads();
#else
    for (int i = tid; i < 16 * KE_; i += 256) {
      int m = i / KE_, c = i % KE_;
      float v = (c < E_) ? xchars[((size_t)(s0 + m) * W_ + tc) * E_ + c] : 0.f;
      Xt[m][c] = (_Float16)v;
    }
    __syncthreads();
#endif

    for (int nt = wave; nt < NT_; nt += 8) {
      // gx = X_t @ Wih^T   (K = 320 -> 10 k-tiles)
      v8f ax = {};
#pragma unroll
      for (int kt = 0; kt < KE_ / 32; ++kt) {
        HFrag a;
        const int m_ = lane & 15, h8_ = ((lane >> 4) & 1) << 3;
#pragma unroll
        for (int r = 0; r < 8; ++r) {
          int k = kt * 32 + ((r >> 2) << 4) + h8_ + ((r & 3) << 1);
          a.u[r] = *reinterpret_cast<const unsigned*>(&Xt[m_][k]);
        }
        v16h b = load_fragB_g(Wih, KE_, nt * 16, kt * 32, lane);
        ax = wmma_f16(a.v, b, ax);
      }
      // gh = h @ Whh^T   (K = 160 -> 5 k-tiles)
      v8f ah = {};
#pragma unroll
      for (int kt = 0; kt < KH_ / 32; ++kt) {
        HFrag a;
        const int m_ = lane & 15, h8_ = ((lane >> 4) & 1) << 3;
#pragma unroll
        for (int r = 0; r < 8; ++r) {
          int k = kt * 32 + ((r >> 2) << 4) + h8_ + ((r & 3) << 1);
          a.u[r] = *reinterpret_cast<const unsigned*>(&Hh[m_][k]);
        }
        v16h b = load_fragB_g(Whh, KH_, nt * 16, kt * 32, lane);
        ah = wmma_f16(a.v, b, ah);
      }
      const int n = nt * 16 + (lane & 15);
      const float bx = (n < N3_) ? bih[n] : 0.f;
      const float bh = (n < N3_) ? bhh[n] : 0.f;
      const int mrow = ((lane >> 4) & 1) * 8;
#pragma unroll
      for (int r = 0; r < 8; ++r) {
        float ghv = ah[r] + bh;
        Gs[mrow + r][n] = ax[r] + bx + ghv;
        if (n >= 2 * H_ && n < N3_) Ghn[mrow + r][n - 2 * H_] = ghv;
      }
    }
    __syncthreads();

    // gate math: r,z,n thirds; tanh(xn + r*hn) = tanh((xn+hn) + (r-1)*hn)
    for (int i = tid; i < 16 * H_; i += 256) {
      int m = i / H_, j = i % H_;
      float r = sigm_(Gs[m][j]);
      float z = sigm_(Gs[m][j + H_]);
      float nn = tanh_(Gs[m][j + 2 * H_] + (r - 1.f) * Ghn[m][j]);
      float hn = (1.f - z) * nn + z * Hf[m][j];
      Hf[m][j] = hn;
      Hh[m][j] = (_Float16)hn;
    }
    __syncthreads();
  }
  for (int i = tid; i < 16 * H_; i += 256) {
    int m = i / H_, j = i % H_;
    h_char[(size_t)(s0 + m) * (2 * H_) + dir * H_ + j] = Hf[m][j];
  }
}

// ---------- sent_in = concat(x_words, h_char) as f16; copy x_words to out ----------
__global__ __launch_bounds__(256)
void k_build_sent(const float* __restrict__ xw, const float* __restrict__ hch,
                  _Float16* __restrict__ si, float* __restrict__ xw_out) {
  const int stride = gridDim.x * blockDim.x;
  const int t0 = blockIdx.x * blockDim.x + threadIdx.x;
  for (int i = t0; i < S_ * KS_; i += stride) {
    int s = i / KS_, c = i % KS_;
    float v = (c < E_) ? xw[(size_t)s * E_ + c]
                       : ((c < E_ + 2 * H_) ? hch[(size_t)s * (2 * H_) + (c - E_)] : 0.f);
    si[i] = (_Float16)v;
  }
  for (int i = t0; i < S_ * E_; i += stride) xw_out[i] = xw[i];
}

// ---------- sentence gates_x GEMM: (S,608) @ (608->450) per direction ----------
__global__ __launch_bounds__(256)
void k_sent_gates(const _Float16* __restrict__ si,
                  const _Float16* __restrict__ WihF, const _Float16* __restrict__ WihB,
                  const float* __restrict__ bihF, const float* __restrict__ bihB,
                  float* __restrict__ gxsF, float* __restrict__ gxsB) {
  const int dir = blockIdx.y;
  const _Float16* Wih = dir ? WihB : WihF;
  const float* bih = dir ? bihB : bihF;
  float* gxs = dir ? gxsB : gxsF;
  const int s0 = blockIdx.x * 16;
  const int lane = threadIdx.x & 31, wave = threadIdx.x >> 5;
  __builtin_prefetch(Wih, 0, 3);

  for (int nt = wave; nt < NT_; nt += 8) {
    v8f acc = {};
#pragma unroll
    for (int kt = 0; kt < KS_ / 32; ++kt) {
      v16h a = load_fragA_g(si, KS_, s0, kt * 32, lane);
      v16h b = load_fragB_g(Wih, KS_, nt * 16, kt * 32, lane);
      acc = wmma_f16(a, b, acc);
    }
    const int n = nt * 16 + (lane & 15);
    const float bb = (n < N3_) ? bih[n] : 0.f;
    const int mrow = ((lane >> 4) & 1) * 8;
#pragma unroll
    for (int r = 0; r < 8; ++r)
      gxs[(size_t)(s0 + mrow + r) * NP_ + n] = acc[r] + bb;
  }
}

// ---------- sequential sentence scan, one workgroup per direction ----------
__global__ __launch_bounds__(1024)
void k_sent_scan(const _Float16* __restrict__ WhhTF, const _Float16* __restrict__ WhhTB,
                 const float* __restrict__ bhhF, const float* __restrict__ bhhB,
                 const float* __restrict__ gxsF, const float* __restrict__ gxsB,
                 float* __restrict__ out) {
  const int dir = blockIdx.x;
  const _Float16* WhhT = dir ? WhhTB : WhhTF;  // (KH_ x NP_) column-major weights
  const float* bhh = dir ? bhhB : bhhF;
  const float* gxs = dir ? gxsB : gxsF;
  const int tid = threadIdx.x;
  const int half = tid >> 9;        // K split: 0 -> k<75, 1 -> k>=75
  const int j = tid & 511;

  __shared__ float hcur[H_];
  __shared__ float part[2][N3_];
  __shared__ float gh[N3_];
  if (tid < H_) hcur[tid] = 0.f;
  if (j < N3_) __builtin_prefetch(&WhhT[half * 75 * NP_ + j], 0, 3);
  __syncthreads();

  for (int t = 0; t < S_; ++t) {
    const int row = dir ? (S_ - 1 - t) : t;
    if (j < N3_) {
      float acc = 0.f;
      const int k0 = half * 75;
#pragma unroll
      for (int k = 0; k < 75; ++k)
        acc += (float)WhhT[(size_t)(k0 + k) * NP_ + j] * hcur[k0 + k];
      part[half][j] = acc;
    }
    __syncthreads();
    if (tid < N3_) gh[tid] = part[0][tid] + part[1][tid] + bhh[tid];
    __syncthreads();
    if (tid < H_) {
      const float* gx = gxs + (size_t)row * NP_;
      float r = sigm_(gx[tid] + gh[tid]);
      float z = sigm_(gx[tid + H_] + gh[tid + H_]);
      float nn = tanh_(gx[tid + 2 * H_] + r * gh[tid + 2 * H_]);
      float hn = (1.f - z) * nn + z * hcur[tid];
      hcur[tid] = hn;
      out[(size_t)row * (2 * H_) + dir * H_ + tid] = hn;
    }
    __syncthreads();
  }
}

extern "C" void kernel_launch(void* const* d_in, const int* in_sizes, int n_in,
                              void* d_out, int out_size, void* d_ws, size_t ws_size,
                              hipStream_t stream) {
  const float* x_words = (const float*)d_in[0];
  const float* x_chars = (const float*)d_in[1];
  const float* cWih_f = (const float*)d_in[2];
  const float* cWhh_f = (const float*)d_in[3];
  const float* cbih_f = (const float*)d_in[4];
  const float* cbhh_f = (const float*)d_in[5];
  const float* cWih_b = (const float*)d_in[6];
  const float* cWhh_b = (const float*)d_in[7];
  const float* cbih_b = (const float*)d_in[8];
  const float* cbhh_b = (const float*)d_in[9];
  const float* sWih_f = (const float*)d_in[10];
  const float* sWhh_f = (const float*)d_in[11];
  const float* sbih_f = (const float*)d_in[12];
  const float* sbhh_f = (const float*)d_in[13];
  const float* sWih_b = (const float*)d_in[14];
  const float* sWhh_b = (const float*)d_in[15];
  const float* sbih_b = (const float*)d_in[16];
  const float* sbhh_b = (const float*)d_in[17];

  char* w = (char*)d_ws;
  auto alloc = [&](size_t bytes) {
    void* p = (void*)w;
    w += (bytes + 255) & ~(size_t)255;
    return p;
  };
  _Float16* wCWihF = (_Float16*)alloc((size_t)NP_ * KE_ * 2);
  _Float16* wCWihB = (_Float16*)alloc((size_t)NP_ * KE_ * 2);
  _Float16* wCWhhF = (_Float16*)alloc((size_t)NP_ * KH_ * 2);
  _Float16* wCWhhB = (_Float16*)alloc((size_t)NP_ * KH_ * 2);
  _Float16* wSWihF = (_Float16*)alloc((size_t)NP_ * KS_ * 2);
  _Float16* wSWihB = (_Float16*)alloc((size_t)NP_ * KS_ * 2);
  _Float16* wSWhhTF = (_Float16*)alloc((size_t)KH_ * NP_ * 2);
  _Float16* wSWhhTB = (_Float16*)alloc((size_t)KH_ * NP_ * 2);
  float*    wHchar  = (float*)alloc((size_t)S_ * 2 * H_ * 4);
  _Float16* wSentIn = (_Float16*)alloc((size_t)S_ * KS_ * 2);
  float*    wGxsF   = (float*)alloc((size_t)S_ * NP_ * 4);
  float*    wGxsB   = (float*)alloc((size_t)S_ * NP_ * 4);

  float* out_u  = (float*)d_out;                       // (S, 2H)
  float* out_xw = (float*)d_out + (size_t)S_ * 2 * H_; // (S, E)

  // 1) weight prep
  k_pad_w<<<256, 256, 0, stream>>>(cWih_f, wCWihF, N3_, E_, KE_, NP_ * KE_);
  k_pad_w<<<256, 256, 0, stream>>>(cWih_b, wCWihB, N3_, E_, KE_, NP_ * KE_);
  k_pad_w<<<128, 256, 0, stream>>>(cWhh_f, wCWhhF, N3_, H_, KH_, NP_ * KH_);
  k_pad_w<<<128, 256, 0, stream>>>(cWhh_b, wCWhhB, N3_, H_, KH_, NP_ * KH_);
  k_pad_w<<<512, 256, 0, stream>>>(sWih_f, wSWihF, N3_, 2 * E_, KS_, NP_ * KS_);
  k_pad_w<<<512, 256, 0, stream>>>(sWih_b, wSWihB, N3_, 2 * E_, KS_, NP_ * KS_);
  k_transpose_w<<<128, 256, 0, stream>>>(sWhh_f, wSWhhTF);
  k_transpose_w<<<128, 256, 0, stream>>>(sWhh_b, wSWhhTB);

  // 2) fused char BiGRU (WMMA + TDM staging) -> h_char (S, 2H)
  dim3 gc(S_ / 16, 2);
  k_char_bigru<<<gc, 256, 0, stream>>>(x_chars, wCWihF, wCWihB, wCWhhF, wCWhhB,
                                       cbih_f, cbih_b, cbhh_f, cbhh_b, wHchar);

  // 3) sent_in = [x_words, h_char] f16; also pass x_words through to output
  k_build_sent<<<512, 256, 0, stream>>>(x_words, wHchar, wSentIn, out_xw);

  // 4) sentence gates_x (WMMA) for both directions
  dim3 gs(S_ / 16, 2);
  k_sent_gates<<<gs, 256, 0, stream>>>(wSentIn, wSWihF, wSWihB, sbih_f, sbih_b,
                                       wGxsF, wGxsB);

  // 5) sequential scan, one persistent WG per direction -> u (S, 2H)
  k_sent_scan<<<2, 1024, 0, stream>>>(wSWhhTF, wSWhhTB, sbhh_f, sbhh_b,
                                      wGxsF, wGxsB, out_u);
}
